// Attention_19189913879065
// MI455X (gfx1250) — compile-verified, matrix operand order checked by reading, not run on hardware
//
#include <hip/hip_runtime.h>
#include <hip/hip_bf16.h>

// ---------------------------------------------------------------------------
// Fused additive-attention kernel for MI455X (gfx1250, wave32, WMMA + TDM).
//
// One workgroup (256 threads = 8 wave32) per batch element b:
//   Phase A: base[a] = (hidden[b] @ W_hidden)[a] + b_hidden[a] + b_feat[a]
//   Phase B: E = features[b] (49x2048) @ W_feat (2048x256) via
//            v_wmma_f32_16x16x32_bf16 (M padded 49->64, fp32 accum).
//            W_feat K-chunks staged by TENSOR_LOAD_TO_LDS (double buffered),
//            then transposed/converted to bf16 [a][k] in LDS.
//            scores[r] = sum_a tanh(E[r,a] + base[a]) * V_w[a]
//            (V_b omitted: softmax is shift-invariant; scores not output)
//   Phase C: alpha = softmax(scores); context[b,f] = sum_r alpha[r]*features
//
// Memory-bound: features = 205 MB dominates (2 passes ~= 18 us @ 23.3 TB/s);
// fusion keeps feat_proj/scores/alpha on-chip, W_feat stays L2-resident.
// ---------------------------------------------------------------------------

typedef __attribute__((ext_vector_type(16))) __bf16        v16bf;
typedef __attribute__((ext_vector_type(8)))  float         v8f;
typedef __attribute__((ext_vector_type(4)))  unsigned int  v4u;
typedef __attribute__((ext_vector_type(8)))  int           v8i;
typedef __attribute__((ext_vector_type(4)))  int           v4i;

#define FDIM 2048
#define ADIM 256
#define RNUM 49
#define KSTEPS (FDIM / 32)   // 64 chunks of K=32

#if defined(__HIP_DEVICE_COMPILE__) && __has_builtin(__builtin_amdgcn_tensor_load_to_lds)
#define HAS_TDM 1
#else
#define HAS_TDM 0
#endif

#if HAS_TDM
// Issue one TDM 2D tile load: 32 rows x 256 cols of fp32 (W_feat chunk) -> LDS.
// D# built per CDNA5 ISA 8.3/8.4 (group0: count/lds_addr/global_addr/type=2;
// group1: data_size=4B, tensor_dim0=256, tile 256x32, dim0_stride=256).
__device__ __forceinline__ void tdm_load_chunk(const float* gsrc, unsigned lds_off) {
    const unsigned long long ga = (unsigned long long)(uintptr_t)gsrc;
    v4u g0;
    g0[0] = 1u;                                   // count=1 (valid user desc)
    g0[1] = lds_off;                              // lds_addr (bytes)
    g0[2] = (unsigned)(ga & 0xffffffffu);         // global_addr[31:0]
    g0[3] = (unsigned)((ga >> 32) & 0x01ffffffu)  // global_addr[56:32]
          | (2u << 30);                           // type = 2 ("image")
    v8i g1;
    g1[0] = (int)(2u << 16);          // data_size=2 (4B), workgroup_mask=0
    g1[1] = (int)(256u << 16);        // tensor_dim0[15:0] -> bits[31:16]
    g1[2] = (int)(2048u << 16);       // tensor_dim0 hi=0 | tensor_dim1 lo (2048)
    g1[3] = (int)(256u << 16);        // tensor_dim1 hi=0 | tile_dim0=256
    g1[4] = 32;                       // tile_dim1=32, tile_dim2=0
    g1[5] = 256;                      // tensor_dim0_stride = 256
    g1[6] = 0;                        // stride hi | tensor_dim1_stride lo
    g1[7] = 0;
    v4i z = {0, 0, 0, 0};
#if defined(__clang_major__) && (__clang_major__ >= 23)
    v8i z8 = {0, 0, 0, 0, 0, 0, 0, 0};
    __builtin_amdgcn_tensor_load_to_lds(g0, g1, z, z, z8, 0);
#else
    __builtin_amdgcn_tensor_load_to_lds(g0, g1, z, z, 0);
#endif
}
#endif

__global__ __launch_bounds__(256)
void attn_fused_kernel(const float* __restrict__ features,  // (B,49,2048)
                       const float* __restrict__ hidden,    // (B,256)
                       const float* __restrict__ W_feat,    // (2048,256)
                       const float* __restrict__ b_feat,    // (256)
                       const float* __restrict__ W_hidden,  // (256,256)
                       const float* __restrict__ b_hidden,  // (256)
                       const float* __restrict__ V_w,       // (256)
                       float* __restrict__ ctx_out,         // (B,2048)
                       float* __restrict__ alpha_out)       // (B,49)
{
    __shared__ __bf16 sB[ADIM][40];   // W_feat chunk, bf16, [a][k], padded pitch
    __shared__ float  sBase[ADIM];
    __shared__ float  sVw[ADIM];
    __shared__ float  sScores[64];
    __shared__ float  sAlpha[64];
#if HAS_TDM
    __shared__ float  sRaw[2][32][ADIM];   // TDM landing buffers (raw fp32)
#endif

    const int b    = blockIdx.x;
    const int tid  = threadIdx.x;
    const int lane = tid & 31;
    const int wave = tid >> 5;

    if (tid < 64) sScores[tid] = 0.0f;

    // ---------------- Phase A: hid_proj + biases (tiny GEMV) ---------------
    {
        float acc = b_hidden[tid] + b_feat[tid];
        const float* hp = hidden + (size_t)b * ADIM;
        #pragma unroll 8
        for (int h = 0; h < ADIM; ++h)
            acc = fmaf(hp[h], W_hidden[h * ADIM + tid], acc);
        sBase[tid] = acc;
        sVw[tid]   = V_w[tid];
    }

#if HAS_TDM
    if (wave == 0)   // one DMA per workgroup: prologue chunk 0 -> buffer 0
        tdm_load_chunk(W_feat, (unsigned)(uintptr_t)(void*)&sRaw[0][0][0]);
#endif

    // ---------------- Phase B: WMMA GEMM -> scores -------------------------
    // wave -> tile map: mt = wave&3 (M rows 16*mt..), nt in [ntbase, ntbase+8)
    const int mt     = wave & 3;
    const int ntbase = (wave >> 2) * 8;
    const int m      = lane & 15;     // M row in A tile / N col in B,C tiles
    const int half   = lane >> 4;     // K-half select per ISA 16-bit layout

    const int  r      = mt * 16 + m;
    const bool rvalid = (r < RNUM);
    const float* arow = features + ((size_t)b * RNUM + (rvalid ? r : 0)) * FDIM;

    v8f acc[8];
    #pragma unroll
    for (int j = 0; j < 8; ++j) acc[j] = (v8f)0.0f;

    for (int kc = 0; kc < KSTEPS; ++kc) {
#if HAS_TDM
        const int cur = kc & 1;
        if (wave == 0)
            __builtin_amdgcn_s_wait_tensorcnt(0);   // buffer[cur] landed
        __syncthreads();   // publish buffer[cur]; prior compute done with sB
        if (wave == 0 && (kc + 1) < KSTEPS)         // prefetch next chunk
            tdm_load_chunk(W_feat + (size_t)(kc + 1) * 32 * ADIM,
                           (unsigned)(uintptr_t)(void*)&sRaw[cur ^ 1][0][0]);
        // transpose + convert: sRaw[cur][k][a] (fp32) -> sB[a][k] (bf16)
        #pragma unroll 8
        for (int kl = 0; kl < 32; ++kl)
            sB[tid][kl] = (__bf16)sRaw[cur][kl][tid];
#else
        __syncthreads();   // prior compute done reading sB
        {
            const float* wcol = W_feat + (size_t)(kc * 32) * ADIM + tid;
            #pragma unroll 8
            for (int kl = 0; kl < 32; ++kl)
                sB[tid][kl] = (__bf16)wcol[kl * ADIM];
        }
#endif
        __syncthreads();

        // A fragment: half 0 holds K 0-7 & 16-23, half 1 holds 8-15 & 24-31
        v16bf afrag;
        const int kbase = kc * 32 + half * 8;
        if (rvalid) {
            const float* p0 = arow + kbase;        // 8 contiguous f32
            const float* p1 = arow + kbase + 16;   // 8 contiguous f32
            #pragma unroll
            for (int i = 0; i < 8; ++i) {
                afrag[i]     = (__bf16)p0[i];
                afrag[i + 8] = (__bf16)p1[i];
            }
            if (kc + 1 < KSTEPS)
                __builtin_prefetch(arow + kbase + 32, 0, 0);  // global_prefetch_b8
        } else {
            #pragma unroll
            for (int i = 0; i < 16; ++i) afrag[i] = (__bf16)0.0f;
        }

        // preload all 8 B fragments, then issue 8 WMMAs back-to-back
        v16bf bfr[8];
        #pragma unroll
        for (int j = 0; j < 8; ++j) {
            const int acol = (ntbase + j) * 16 + m;
            const __bf16* q0 = &sB[acol][half * 8];
            const __bf16* q1 = &sB[acol][16 + half * 8];
            #pragma unroll
            for (int i = 0; i < 8; ++i) {
                bfr[j][i]     = q0[i];
                bfr[j][i + 8] = q1[i];
            }
        }
        #pragma unroll
        for (int j = 0; j < 8; ++j)
            acc[j] = __builtin_amdgcn_wmma_f32_16x16x32_bf16(
                false, afrag, false, bfr[j], (short)0, acc[j], false, false);
    }

    // epilogue: tanh(E + base) * V_w, reduce over a into scores[r]
    // C/D layout: lane n = lane&15, VGPR v holds (row = v + 8*half, col n)
    {
        float sp[8];
        #pragma unroll
        for (int v = 0; v < 8; ++v) sp[v] = 0.0f;

        #pragma unroll
        for (int j = 0; j < 8; ++j) {
            const int acol = (ntbase + j) * 16 + m;
            const float bb = sBase[acol];
            const float vw = sVw[acol];
            #pragma unroll
            for (int v = 0; v < 8; ++v) {
                const float e = tanhf(acc[j][v] + bb);
                sp[v] = fmaf(e, vw, sp[v]);
            }
        }
        // sum across the 16 lanes of each half (N direction)
        #pragma unroll
        for (int v = 0; v < 8; ++v) {
            #pragma unroll
            for (int mask = 1; mask < 16; mask <<= 1)
                sp[v] += __shfl_xor(sp[v], mask, 32);
        }
        if (m == 0) {
            const int rr = mt * 16 + 8 * half;
            #pragma unroll
            for (int v = 0; v < 8; ++v)
                atomicAdd(&sScores[rr + v], sp[v]);   // ds_add_f32
        }
    }
    __syncthreads();

    // ---------------- softmax over 49 scores -------------------------------
    if (tid == 0) {
        float mx = -3.402823466e+38f;
        for (int i = 0; i < RNUM; ++i) mx = fmaxf(mx, sScores[i]);
        float sum = 0.0f;
        for (int i = 0; i < RNUM; ++i) {
            const float ex = __expf(sScores[i] - mx);
            sAlpha[i] = ex;
            sum += ex;
        }
        const float inv = 1.0f / sum;
        for (int i = 0; i < RNUM; ++i) sAlpha[i] *= inv;
    }
    __syncthreads();

    if (tid < RNUM) alpha_out[(size_t)b * RNUM + tid] = sAlpha[tid];

    // ---------------- Phase C: context = alpha . features ------------------
    {
        const float4* fb = reinterpret_cast<const float4*>(
            features + (size_t)b * RNUM * FDIM);
        float4* cb = reinterpret_cast<float4*>(ctx_out + (size_t)b * FDIM);
        #pragma unroll
        for (int c = tid; c < FDIM / 4; c += 256) {
            float4 a4 = make_float4(0.f, 0.f, 0.f, 0.f);
            for (int rr = 0; rr < RNUM; ++rr) {
                const float  al = sAlpha[rr];
                const float4 v  = fb[rr * (FDIM / 4) + c];
                a4.x = fmaf(al, v.x, a4.x);
                a4.y = fmaf(al, v.y, a4.y);
                a4.z = fmaf(al, v.z, a4.z);
                a4.w = fmaf(al, v.w, a4.w);
            }
            cb[c] = a4;
        }
    }
}

extern "C" void kernel_launch(void* const* d_in, const int* in_sizes, int n_in,
                              void* d_out, int out_size, void* d_ws, size_t ws_size,
                              hipStream_t stream) {
    const float* features = (const float*)d_in[0];
    const float* hidden   = (const float*)d_in[1];
    const float* W_feat   = (const float*)d_in[2];
    const float* b_feat   = (const float*)d_in[3];
    const float* W_hidden = (const float*)d_in[4];
    const float* b_hidden = (const float*)d_in[5];
    const float* V_w      = (const float*)d_in[6];
    // d_in[7] = V_b : irrelevant to alpha/context (softmax shift invariance)

    const int B = in_sizes[0] / (RNUM * FDIM);   // 512
    float* ctx   = (float*)d_out;                      // (B, 2048) first
    float* alpha = (float*)d_out + (size_t)B * FDIM;   // (B, 49) second

    attn_fused_kernel<<<B, 256, 0, stream>>>(
        features, hidden, W_feat, b_feat, W_hidden, b_hidden, V_w, ctx, alpha);
}
